// SiLKVGG_35167192219891
// MI455X (gfx1250) — matrix operand order repaired and checked by reading, not compiled, not run on hardware
//
#include <hip/hip_runtime.h>
#include <stdint.h>

typedef __attribute__((ext_vector_type(16))) _Float16 v16h;
typedef __attribute__((ext_vector_type(8)))  float    v8f;

static constexpr int kH = 384, kW = 384, kB = 2;
static constexpr int kHW  = kH * kW;          // 147456
static constexpr int kPIX = kB * kHW;         // 294912
static constexpr int kTOPK = 500;
static constexpr int kCAP  = 8192;
#define INT_MAX_C 0x7fffffff

// ---------------------------------------------------------------------------
// Weight repack: OIHW f32 -> [O][9][I] f16  (tap-major, cin contiguous)
// ---------------------------------------------------------------------------
__global__ void wconv_kernel(const float* __restrict__ src, _Float16* __restrict__ dst,
                             int O, int I) {
    int t = blockIdx.x * blockDim.x + threadIdx.x;
    int total = O * I * 9;
    if (t >= total) return;
    int tap = t % 9; int rest = t / 9;
    int i = rest % I; int o = rest / I;
    dst[(size_t)o * 9 * I + tap * I + i] = (_Float16)src[((size_t)o * I + i) * 9 + tap];
}

// Zero the per-launch state: candidate counters + the 256B zero-pad row used
// by out-of-bounds conv taps (workspace is poisoned, so re-zero every call).
__global__ void init_kernel(int* cnt, _Float16* zpad) {
    if (threadIdx.x < kB) cnt[threadIdx.x] = 0;
    zpad[threadIdx.x] = (_Float16)0.f;     // 128 halves = 256 B
}

// ---------------------------------------------------------------------------
// conv1: 1 -> 64, plain VALU (K=9, too small for WMMA), out NHWC fp16
// ---------------------------------------------------------------------------
__global__ void conv1_kernel(const float* __restrict__ img, const float* __restrict__ w,
                             const float* __restrict__ bias, _Float16* __restrict__ out) {
    int t = blockIdx.x * blockDim.x + threadIdx.x;   // pix*64 + c
    if (t >= kPIX * 64) return;
    int c = t & 63; int pix = t >> 6;
    int x = pix % kW; int by = pix / kW; int y = by % kH;
    float s = bias[c];
    #pragma unroll
    for (int ky = 0; ky < 3; ++ky)
        #pragma unroll
        for (int kx = 0; kx < 3; ++kx) {
            int yy = y + ky - 1, xx = x + kx - 1;
            if ((unsigned)yy < (unsigned)kH && (unsigned)xx < (unsigned)kW)
                s += img[(size_t)(by + ky - 1) * kW + xx] * w[c * 9 + ky * 3 + kx];
        }
    out[(size_t)pix * 64 + c] = (_Float16)(s > 0.f ? s : 0.f);
}

// ---------------------------------------------------------------------------
// 3x3 conv as implicit GEMM on v_wmma_f32_16x16x32_f16.
//   Wave tile: M=16 output channels x N=64 pixels (4 WMMA N-tiles, one row).
//   Per K-step: A loaded once (2x b128), 4x B fragments (2x b128 each,
//   pointer-selected against a zero pad -> no EXEC-mask branches), 4 WMMAs.
//   act : NHWC fp16, w16 : [COUT][9][CIN] fp16, out : NHWC fp16 (bias+ReLU).
// ---------------------------------------------------------------------------
template<int CIN, int COUT>
__global__ __launch_bounds__(32)
void conv3x3_wmma(const _Float16* __restrict__ act, const _Float16* __restrict__ w16,
                  const float* __restrict__ bias, const _Float16* __restrict__ zpad,
                  _Float16* __restrict__ out) {
    const int lane = threadIdx.x;
    const int nl = lane & 15;      // N index within tile / M index for A rows
    const int hf = lane >> 4;      // half select
    const int pix0 = blockIdx.x * 64;            // aligned 64-pixel row segment
    const int m0   = blockIdx.y * 16;
    const int x0  = pix0 % kW;                   // multiple of 64
    const int by  = pix0 / kW;                   // b*kH + y (same for all 64)
    const int y   = by % kH;

    v8f acc0 = {}, acc1 = {}, acc2 = {}, acc3 = {};
    constexpr int NK = (9 * CIN) / 32;
    const _Float16* wrowBase = w16 + (size_t)(m0 + nl) * (9 * CIN);

    #pragma unroll 1
    for (int kb = 0; kb < NK; ++kb) {
        const int k0   = kb * 32;
        const int tap  = k0 / CIN;       // 32-chunk never straddles taps (CIN%32==0)
        const int cin0 = k0 % CIN;
        const int dy = tap / 3 - 1, dx = tap % 3 - 1;
        const int yy = y + dy;
        const bool rowok = (unsigned)yy < (unsigned)kH;

        // ---- A fragment (16x32 f16): two contiguous 16B chunks per lane ----
        union { v16h h; uint4 q[2]; } Am;
        const _Float16* wr = wrowBase + tap * CIN + cin0;
        Am.q[0] = *reinterpret_cast<const uint4*>(wr + hf * 8);
        Am.q[1] = *reinterpret_cast<const uint4*>(wr + 16 + hf * 8);

        // ---- 4x B fragments (32x16 f16): two 16B chunks per lane, same k order as A ----
        const size_t rowBase = ((size_t)(by + dy) * kW) * CIN + cin0;
        union { v16h h; uint4 q[2]; } Bm[4];
        #pragma unroll
        for (int t = 0; t < 4; ++t) {
            const int xx = x0 + t * 16 + nl + dx;
            const bool inb = rowok && ((unsigned)xx < (unsigned)kW);
            const _Float16* bp = inb ? (act + rowBase + (size_t)xx * CIN) : zpad;
            __builtin_prefetch(bp + CIN * kW, 0, 3);   // next-row hint
            Bm[t].q[0] = *reinterpret_cast<const uint4*>(bp + hf * 8);
            Bm[t].q[1] = *reinterpret_cast<const uint4*>(bp + 16 + hf * 8);
        }
        acc0 = __builtin_amdgcn_wmma_f32_16x16x32_f16(false, Am.h, false, Bm[0].h, (short)0, acc0, false, false);
        acc1 = __builtin_amdgcn_wmma_f32_16x16x32_f16(false, Am.h, false, Bm[1].h, (short)0, acc1, false, false);
        acc2 = __builtin_amdgcn_wmma_f32_16x16x32_f16(false, Am.h, false, Bm[2].h, (short)0, acc2, false, false);
        acc3 = __builtin_amdgcn_wmma_f32_16x16x32_f16(false, Am.h, false, Bm[3].h, (short)0, acc3, false, false);
    }

    // D layout: VGPR r, lane -> (m = hf*8 + r, n = nl). Bias + ReLU, pack 8 f16.
    float bv[8];
    #pragma unroll
    for (int r = 0; r < 8; ++r) bv[r] = bias[m0 + hf * 8 + r];
    v8f* accs[4] = { &acc0, &acc1, &acc2, &acc3 };
    #pragma unroll
    for (int t = 0; t < 4; ++t) {
        union { _Float16 h[8]; uint4 q; } st;
        #pragma unroll
        for (int r = 0; r < 8; ++r) {
            float v = (*accs[t])[r] + bv[r];
            st.h[r] = (_Float16)(v > 0.f ? v : 0.f);
        }
        const int pix = pix0 + t * 16 + nl;
        *reinterpret_cast<uint4*>(out + (size_t)pix * COUT + m0 + hf * 8) = st.q;
    }
}

// ---------------------------------------------------------------------------
// Detector 1x1 head: 128 -> 1, + sigmoid => prob[B*HW]
// ---------------------------------------------------------------------------
__global__ void det_head_kernel(const _Float16* __restrict__ dact, const float* __restrict__ w,
                                const float* __restrict__ b2, float* __restrict__ prob) {
    int pix = blockIdx.x * blockDim.x + threadIdx.x;
    if (pix >= kPIX) return;
    const _Float16* dp = dact + (size_t)pix * 128;
    float z = b2[0];
    #pragma unroll 4
    for (int c = 0; c < 128; ++c) z += (float)dp[c] * w[c];
    prob[pix] = 1.0f / (1.0f + expf(-z));
}

// Separable 9x9 max: row pass
__global__ void rowmax_kernel(const float* __restrict__ prob, float* __restrict__ rowm) {
    int pix = blockIdx.x * blockDim.x + threadIdx.x;
    if (pix >= kPIX) return;
    int x = pix % kW;
    float m = prob[pix];
    #pragma unroll
    for (int d = -4; d <= 4; ++d) {
        int xx = x + d;
        if ((unsigned)xx < (unsigned)kW) m = fmaxf(m, prob[pix - x + xx]);
    }
    rowm[pix] = m;
}

// Column pass + threshold/border/NMS test + candidate compaction
__global__ void nms_kernel(const float* __restrict__ prob, const float* __restrict__ rowm,
                           int* __restrict__ cnt, float* __restrict__ candv,
                           int* __restrict__ candi) {
    int pix = blockIdx.x * blockDim.x + threadIdx.x;
    if (pix >= kPIX) return;
    int x = pix % kW; int by = pix / kW; int y = by % kH; int b = by / kH;
    float m = rowm[pix];
    #pragma unroll
    for (int d = -4; d <= 4; ++d) {
        int yy = y + d;
        if ((unsigned)yy < (unsigned)kH) m = fmaxf(m, rowm[pix + d * kW]);
    }
    float p = prob[pix];
    bool border = (y >= 4) && (y < kH - 4) && (x >= 4) && (x < kW - 4);
    if (p >= m && p > 0.8f && border) {
        int pos = atomicAdd(&cnt[b], 1);
        if (pos < kCAP) { candv[b * kCAP + pos] = p; candi[b * kCAP + pos] = y * kW + x; }
    }
}

// Deterministic per-image top-500 (value desc, index asc tiebreak); zero-pad
// with ascending indices like jax.lax.top_k over an all-zero remainder.
__global__ __launch_bounds__(256)
void topk_kernel(const int* __restrict__ cnt, float* __restrict__ candv,
                 const int* __restrict__ candi, int* __restrict__ topi,
                 float* __restrict__ pos_out) {
    __shared__ float sv[256]; __shared__ int sid[256]; __shared__ int spos[256];
    const int b = blockIdx.x, tid = threadIdx.x;
    int n = cnt[b]; if (n > kCAP) n = kCAP;
    float* cv = candv + b * kCAP;
    const int* ci = candi + b * kCAP;
    int nsel = 0;
    for (int k = 0; k < kTOPK; ++k) {
        float bv = -1.f; int bi = INT_MAX_C; int bp = -1;
        for (int j = tid; j < n; j += 256) {
            float v = cv[j]; int id = ci[j];
            if (v > bv || (v == bv && id < bi)) { bv = v; bi = id; bp = j; }
        }
        sv[tid] = bv; sid[tid] = bi; spos[tid] = bp;
        __syncthreads();
        for (int s = 128; s > 0; s >>= 1) {
            if (tid < s) {
                float v2 = sv[tid + s]; int id2 = sid[tid + s];
                if (v2 > sv[tid] || (v2 == sv[tid] && id2 < sid[tid])) {
                    sv[tid] = v2; sid[tid] = id2; spos[tid] = spos[tid + s];
                }
            }
            __syncthreads();
        }
        if (tid == 0) {
            int idx; float vv;
            if (sv[0] > 0.f) { idx = sid[0]; vv = sv[0]; cv[spos[0]] = -1.f; ++nsel; }
            else             { idx = k - nsel; vv = 0.f; }   // zero-padded slots
            topi[b * kTOPK + k] = idx;
            float* pr = pos_out + ((size_t)b * kTOPK + k) * 3;
            pr[0] = (float)(idx / kW); pr[1] = (float)(idx % kW); pr[2] = vv;
        }
        __syncthreads();
    }
}

// Descriptor 1x1 projection (256->128) ONLY at keypoints, + L2 normalize.
__global__ __launch_bounds__(128)
void desc_kernel(const _Float16* __restrict__ e, const float* __restrict__ w2,
                 const float* __restrict__ b2, const int* __restrict__ topi,
                 const float* __restrict__ scale, float* __restrict__ desc_out) {
    __shared__ float red[128];
    const int blk = blockIdx.x;            // b*500 + k
    const int b = blk / kTOPK;
    const int idx = topi[blk];
    const int d = threadIdx.x;
    const _Float16* ep = e + ((size_t)b * kHW + idx) * 256;
    const float* wp = w2 + d * 256;
    float s = b2[d];
    #pragma unroll 4
    for (int i = 0; i < 256; ++i) s += (float)ep[i] * wp[i];
    red[d] = s * s;
    __syncthreads();
    for (int st = 64; st > 0; st >>= 1) { if (d < st) red[d] += red[d + st]; __syncthreads(); }
    float norm = fmaxf(sqrtf(red[0]), 1e-12f);
    desc_out[(size_t)blk * 128 + d] = scale[0] * s / norm;
}

// ---------------------------------------------------------------------------
extern "C" void kernel_launch(void* const* d_in, const int* in_sizes, int n_in,
                              void* d_out, int out_size, void* d_ws, size_t ws_size,
                              hipStream_t stream) {
    (void)in_sizes; (void)n_in; (void)out_size; (void)ws_size;
    const float* images = (const float*)d_in[0];
    const float* bb_w1 = (const float*)d_in[1];  const float* bb_b1 = (const float*)d_in[2];
    const float* bb_w2 = (const float*)d_in[3];  const float* bb_b2 = (const float*)d_in[4];
    const float* bb_w3 = (const float*)d_in[5];  const float* bb_b3 = (const float*)d_in[6];
    const float* bb_w4 = (const float*)d_in[7];  const float* bb_b4 = (const float*)d_in[8];
    const float* det_w1 = (const float*)d_in[9]; const float* det_b1 = (const float*)d_in[10];
    const float* det_w2 = (const float*)d_in[11];const float* det_b2 = (const float*)d_in[12];
    const float* des_w1 = (const float*)d_in[13];const float* des_b1 = (const float*)d_in[14];
    const float* des_w2 = (const float*)d_in[15];const float* des_b2 = (const float*)d_in[16];
    const float* scalef = (const float*)d_in[17];

    uint8_t* ws = (uint8_t*)d_ws;
    size_t off = 0;
    auto alloc = [&](size_t n) { size_t o = off; off = (off + n + 255) & ~(size_t)255; return o; };

    const size_t o_zp  = alloc(256);                      // zero pad row
    const size_t o_w2  = alloc((size_t)64  * 9 * 64  * 2);
    const size_t o_w3  = alloc((size_t)128 * 9 * 64  * 2);
    const size_t o_w4  = alloc((size_t)128 * 9 * 128 * 2);
    const size_t o_wd1 = alloc((size_t)128 * 9 * 128 * 2);
    const size_t o_we1 = alloc((size_t)256 * 9 * 128 * 2);
    const size_t o_a1  = alloc((size_t)kPIX * 64  * 2);   // conv1 out
    const size_t o_a2  = alloc((size_t)kPIX * 64  * 2);   // conv2 out
    const size_t o_a3  = alloc((size_t)kPIX * 128 * 2);   // conv3 out
    const size_t o_a4  = alloc((size_t)kPIX * 128 * 2);   // feats
    const size_t o_ad  = alloc((size_t)kPIX * 128 * 2);   // det1 out
    const size_t o_ae  = alloc((size_t)kPIX * 256 * 2);   // des1 out
    const size_t o_pr  = alloc((size_t)kPIX * 4);         // prob
    const size_t o_rm  = alloc((size_t)kPIX * 4);         // rowmax
    const size_t o_cv  = alloc((size_t)kB * kCAP * 4);
    const size_t o_ci  = alloc((size_t)kB * kCAP * 4);
    const size_t o_cnt = alloc(8);
    const size_t o_ti  = alloc((size_t)kB * kTOPK * 4);

    _Float16* zpad = (_Float16*)(ws + o_zp);
    _Float16* w2_16 = (_Float16*)(ws + o_w2); _Float16* w3_16 = (_Float16*)(ws + o_w3);
    _Float16* w4_16 = (_Float16*)(ws + o_w4); _Float16* wd1_16 = (_Float16*)(ws + o_wd1);
    _Float16* we1_16 = (_Float16*)(ws + o_we1);
    _Float16* a1 = (_Float16*)(ws + o_a1); _Float16* a2 = (_Float16*)(ws + o_a2);
    _Float16* a3 = (_Float16*)(ws + o_a3); _Float16* a4 = (_Float16*)(ws + o_a4);
    _Float16* ad = (_Float16*)(ws + o_ad); _Float16* ae = (_Float16*)(ws + o_ae);
    float* prob = (float*)(ws + o_pr); float* rowm = (float*)(ws + o_rm);
    float* candv = (float*)(ws + o_cv); int* candi = (int*)(ws + o_ci);
    int* cnt = (int*)(ws + o_cnt); int* topi = (int*)(ws + o_ti);

    float* desc_out = (float*)d_out;                    // [2,500,128]
    float* pos_out  = (float*)d_out + kB * kTOPK * 128; // [2,500,3]

    // Weight repacks + per-launch init
    auto wgrid = [](int O, int I) { return dim3((O * I * 9 + 255) / 256); };
    wconv_kernel<<<wgrid(64, 64),  256, 0, stream>>>(bb_w2,  w2_16, 64, 64);
    wconv_kernel<<<wgrid(128, 64), 256, 0, stream>>>(bb_w3,  w3_16, 128, 64);
    wconv_kernel<<<wgrid(128,128), 256, 0, stream>>>(bb_w4,  w4_16, 128, 128);
    wconv_kernel<<<wgrid(128,128), 256, 0, stream>>>(det_w1, wd1_16, 128, 128);
    wconv_kernel<<<wgrid(256,128), 256, 0, stream>>>(des_w1, we1_16, 256, 128);
    init_kernel<<<1, 128, 0, stream>>>(cnt, zpad);

    // Backbone
    conv1_kernel<<<(kPIX * 64 + 255) / 256, 256, 0, stream>>>(images, bb_w1, bb_b1, a1);
    conv3x3_wmma<64, 64>  <<<dim3(kPIX / 64, 4),  32, 0, stream>>>(a1, w2_16,  bb_b2,  zpad, a2);
    conv3x3_wmma<64, 128> <<<dim3(kPIX / 64, 8),  32, 0, stream>>>(a2, w3_16,  bb_b3,  zpad, a3);
    conv3x3_wmma<128,128> <<<dim3(kPIX / 64, 8),  32, 0, stream>>>(a3, w4_16,  bb_b4,  zpad, a4);
    // Heads
    conv3x3_wmma<128,128> <<<dim3(kPIX / 64, 8),  32, 0, stream>>>(a4, wd1_16, det_b1, zpad, ad);
    conv3x3_wmma<128,256> <<<dim3(kPIX / 64, 16), 32, 0, stream>>>(a4, we1_16, des_b1, zpad, ae);

    // Detector post-processing
    det_head_kernel<<<(kPIX + 255) / 256, 256, 0, stream>>>(ad, det_w2, det_b2, prob);
    rowmax_kernel<<<(kPIX + 255) / 256, 256, 0, stream>>>(prob, rowm);
    nms_kernel<<<(kPIX + 255) / 256, 256, 0, stream>>>(prob, rowm, cnt, candv, candi);
    topk_kernel<<<kB, 256, 0, stream>>>(cnt, candv, candi, topi, pos_out);

    // Sparse descriptors
    desc_kernel<<<kB * kTOPK, 128, 0, stream>>>(ae, des_w2, des_b2, topi, scalef, desc_out);
}
